// MLPAttention_12567074308787
// MI455X (gfx1250) — matrix operand order repaired and checked by reading, not compiled, not run on hardware
//
#include <hip/hip_runtime.h>
#include <hip/hip_bf16.h>

typedef __attribute__((ext_vector_type(16))) _Float16 v16h;
typedef __attribute__((ext_vector_type(8)))  _Float16 v8h;
typedef __attribute__((ext_vector_type(8)))  float    v8f;

#define SRC_LEN  1024
#define BSZ      64
#define CTX_DIM  1024
#define ATTN_DIM 1024
#define NEG_MASK (-1000000.0f)
#define LDSK     1032   // 1024 + 8 halfs padding -> row stride 2064B, spreads LDS banks

// gfx1250 hardware tanh (TRANS op, co-executes with VALU). Fallback to inline
// asm v_tanh_f32, which the gfx1250 assembler accepts per the CDNA5 TRANS set.
__device__ __forceinline__ float fast_tanh(float x) {
#if __has_builtin(__builtin_amdgcn_tanhf)
    return __builtin_amdgcn_tanhf(x);
#else
    float r;
    asm("v_tanh_f32 %0, %1" : "=v"(r) : "v"(x));
    return r;
#endif
}

// ---------------------------------------------------------------------------
// Kernel A: decfull[b][a] = decoder_state[b,:]·W_dec[a,:] + b_dec[a] + b_enc[a]
// ---------------------------------------------------------------------------
__global__ void dec_proj_kernel(const float* __restrict__ dec_state,
                                const float* __restrict__ W_dec,
                                const float* __restrict__ b_dec,
                                const float* __restrict__ b_enc,
                                float* __restrict__ decfull) {
    int idx = blockIdx.x * blockDim.x + threadIdx.x;   // 65536 threads
    int b = idx & (BSZ - 1);
    int a = idx >> 6;
    const float4* xr = (const float4*)(dec_state + (size_t)b * CTX_DIM);
    const float4* wr = (const float4*)(W_dec + (size_t)a * CTX_DIM);
    float acc = b_dec[a] + b_enc[a];
    for (int c = 0; c < CTX_DIM / 4; ++c) {
        float4 x = xr[c], w = wr[c];
        acc += x.x * w.x + x.y * w.y + x.z * w.z + x.w * w.w;
    }
    decfull[(size_t)b * ATTN_DIM + a] = acc;
}

// ---------------------------------------------------------------------------
// Kernel W: W_enc f32 -> f16
// ---------------------------------------------------------------------------
__global__ void convert_w_kernel(const float* __restrict__ W,
                                 _Float16* __restrict__ W16, int n) {
    int i = blockIdx.x * blockDim.x + threadIdx.x;
    if (i < n) W16[i] = (_Float16)W[i];
}

// ---------------------------------------------------------------------------
// Kernel B: fused enc-GEMM + tanh + v_score dot. One workgroup per s.
//   16 waves; wave w owns columns [w*64, w*64+64); 4 row-tiles x 4 col-tiles.
// ---------------------------------------------------------------------------
__launch_bounds__(512, 1)
__global__ void attn_score_kernel(const float* __restrict__ src_hids,
                                  const _Float16* __restrict__ W16,
                                  const float* __restrict__ decfull,
                                  const float* __restrict__ v_score,
                                  float* __restrict__ scores_ws /* [b][s] */) {
    extern __shared__ char smem[];
    _Float16* X = (_Float16*)smem;                       // [64][LDSK] f16
    float* scoreAcc = (float*)(smem + (size_t)BSZ * LDSK * sizeof(_Float16));

    const int s   = blockIdx.x;
    const int tid = threadIdx.x;

    if (tid < BSZ) scoreAcc[tid] = 0.0f;

    // ---- stage source_hids[s] (64x1024 f32) into LDS as f16 ----
    const float4* gsrc = (const float4*)(src_hids + (size_t)s * BSZ * CTX_DIM);
    for (int i = tid; i < BSZ * (CTX_DIM / 4); i += 512) {
        int row = i >> 8;          // /256 float4 per row
        int c4  = i & 255;
        float4 v = gsrc[i];
        _Float16* dst = X + (size_t)row * LDSK + c4 * 4;
        dst[0] = (_Float16)v.x; dst[1] = (_Float16)v.y;
        dst[2] = (_Float16)v.z; dst[3] = (_Float16)v.w;
    }
    __syncthreads();

    const int wave = tid >> 5;        // 0..15
    const int lane = tid & 31;
    const int n16  = lane & 15;
    const int half = lane >> 4;
    const int col_base = wave * 64;

    v8f acc[4][4] = {};               // 16 accumulator tiles (f32)

    for (int kb = 0; kb < CTX_DIM; kb += 32) {
        // A fragments: f16 16x32, lane holds row m=n16, 16 contiguous K at k0 = 16*half
        v16h afrag[4];
#pragma unroll
        for (int rt = 0; rt < 4; ++rt) {
            const _Float16* p = X + (size_t)(rt * 16 + n16) * LDSK + kb + half * 16;
            v8h lo = *(const v8h*)p;
            v8h hi = *(const v8h*)(p + 8);
            afrag[rt] = __builtin_shufflevector(lo, hi,
                0, 1, 2, 3, 4, 5, 6, 7, 8, 9, 10, 11, 12, 13, 14, 15);
        }
        // B fragments: lane holds col n=n16, 16 contiguous K at k0 = 16*half
        v16h bfrag[4];
#pragma unroll
        for (int ct = 0; ct < 4; ++ct) {
            const _Float16* q = W16 +
                (size_t)(col_base + ct * 16 + n16) * CTX_DIM + kb + half * 16;
            bfrag[ct] = *(const v16h*)q;
        }
#pragma unroll
        for (int rt = 0; rt < 4; ++rt) {
#pragma unroll
            for (int ct = 0; ct < 4; ++ct) {
                acc[rt][ct] = __builtin_amdgcn_wmma_f32_16x16x32_f16(
                    false, afrag[rt], false, bfrag[ct],
                    (short)0, acc[rt][ct], false, false);
            }
        }
    }

    // ---- epilogue: tanh(enc + decfull) · v_score, reduce over a ----
    float vs[4];
#pragma unroll
    for (int ct = 0; ct < 4; ++ct) vs[ct] = v_score[col_base + ct * 16 + n16];

#pragma unroll
    for (int rt = 0; rt < 4; ++rt) {
#pragma unroll
        for (int i = 0; i < 8; ++i) {
            int b = rt * 16 + half * 8 + i;   // C/D layout: m = vgpr + 8*(lane/16)
            float val = 0.0f;
#pragma unroll
            for (int ct = 0; ct < 4; ++ct) {
                int a = col_base + ct * 16 + n16;
                float e = acc[rt][ct][i] + decfull[(size_t)b * ATTN_DIM + a];
                val += fast_tanh(e) * vs[ct];
            }
            // reduce the 16 lanes of this half-wave (xor<16 stays in half)
            val += __shfl_xor(val, 1, 32);
            val += __shfl_xor(val, 2, 32);
            val += __shfl_xor(val, 4, 32);
            val += __shfl_xor(val, 8, 32);
            if (n16 == 0) atomicAdd(&scoreAcc[b], val);
        }
    }
    __syncthreads();
    if (tid < BSZ) scores_ws[(size_t)tid * SRC_LEN + s] = scoreAcc[tid];
}

// ---------------------------------------------------------------------------
// Kernel C: masked softmax over s, per b. Writes attn [s][b] (output) and
// attn_ws [b][s] (for the ctx pass).
// ---------------------------------------------------------------------------
__global__ void softmax_kernel(const float* __restrict__ scores_ws, // [b][s]
                               const int* __restrict__ src_lengths,
                               float* __restrict__ attn_out,        // [s][b]
                               float* __restrict__ attn_ws) {       // [b][s]
    __shared__ float red[256];
    int b = blockIdx.x, tid = threadIdx.x;
    int len = src_lengths[b];
    const float* row = scores_ws + (size_t)b * SRC_LEN;

    float v[4];
    float mx = -3.4e38f;
#pragma unroll
    for (int j = 0; j < 4; ++j) {
        int sidx = tid + j * 256;
        float x = (sidx < len) ? row[sidx] : NEG_MASK;
        v[j] = x;
        mx = fmaxf(mx, x);
    }
    red[tid] = mx; __syncthreads();
    for (int off = 128; off > 0; off >>= 1) {
        if (tid < off) red[tid] = fmaxf(red[tid], red[tid + off]);
        __syncthreads();
    }
    mx = red[0]; __syncthreads();

    float sum = 0.0f;
#pragma unroll
    for (int j = 0; j < 4; ++j) { v[j] = expf(v[j] - mx); sum += v[j]; }
    red[tid] = sum; __syncthreads();
    for (int off = 128; off > 0; off >>= 1) {
        if (tid < off) red[tid] += red[tid + off];
        __syncthreads();
    }
    float inv = 1.0f / red[0];
#pragma unroll
    for (int j = 0; j < 4; ++j) {
        int sidx = tid + j * 256;
        float a = v[j] * inv;
        attn_out[(size_t)sidx * BSZ + b] = a;
        attn_ws[(size_t)b * SRC_LEN + sidx] = a;
    }
}

// ---------------------------------------------------------------------------
// Kernel Z + D: ctx[b][c] = sum_s source_hids[s][b][c] * attn[s][b]
// ---------------------------------------------------------------------------
__global__ void zero_kernel(float* __restrict__ p, int n) {
    int i = blockIdx.x * blockDim.x + threadIdx.x;
    if (i < n) p[i] = 0.0f;
}

__global__ void ctx_kernel(const float* __restrict__ src_hids,
                           const float* __restrict__ attn_ws, // [b][s]
                           float* __restrict__ ctx) {         // [b][c]
    int b  = blockIdx.x;        // 64
    int sc = blockIdx.y;        // 8 s-chunks of 128
    int t  = threadIdx.x;       // 256 threads x float4 = 1024 cols
    float4 acc = make_float4(0.f, 0.f, 0.f, 0.f);
    const float4* base = (const float4*)src_hids;
    int s0 = sc * 128;
    for (int s = s0; s < s0 + 128; ++s) {
        float w = attn_ws[(size_t)b * SRC_LEN + s];
        float4 h = base[(size_t)(s * BSZ + b) * (CTX_DIM / 4) + t];
        acc.x += w * h.x; acc.y += w * h.y; acc.z += w * h.z; acc.w += w * h.w;
    }
    float* o = ctx + (size_t)b * CTX_DIM + t * 4;
    atomicAdd(o + 0, acc.x); atomicAdd(o + 1, acc.y);
    atomicAdd(o + 2, acc.z); atomicAdd(o + 3, acc.w);
}

// ---------------------------------------------------------------------------
extern "C" void kernel_launch(void* const* d_in, const int* in_sizes, int n_in,
                              void* d_out, int out_size, void* d_ws, size_t ws_size,
                              hipStream_t stream) {
    const float* dec_state = (const float*)d_in[0];
    const float* src_hids  = (const float*)d_in[1];
    const int*   src_len   = (const int*)d_in[2];
    const float* W_enc     = (const float*)d_in[3];
    const float* b_enc     = (const float*)d_in[4];
    const float* W_dec     = (const float*)d_in[5];
    const float* b_dec     = (const float*)d_in[6];
    const float* v_score   = (const float*)d_in[7];

    float* out_ctx  = (float*)d_out;                 // [64][1024]
    float* out_attn = out_ctx + BSZ * CTX_DIM;       // [1024][64]

    char* ws = (char*)d_ws;
    float*    decfull   = (float*)ws;     ws += (size_t)BSZ * ATTN_DIM * sizeof(float);
    _Float16* W16       = (_Float16*)ws;  ws += (size_t)ATTN_DIM * CTX_DIM * sizeof(_Float16);
    float*    scores_ws = (float*)ws;     ws += (size_t)BSZ * SRC_LEN * sizeof(float);
    float*    attn_ws   = (float*)ws;     ws += (size_t)BSZ * SRC_LEN * sizeof(float);

    dec_proj_kernel<<<(BSZ * ATTN_DIM) / 256, 256, 0, stream>>>(
        dec_state, W_dec, b_dec, b_enc, decfull);
    convert_w_kernel<<<(ATTN_DIM * CTX_DIM) / 256, 256, 0, stream>>>(
        W_enc, W16, ATTN_DIM * CTX_DIM);

    size_t lds_bytes = (size_t)BSZ * LDSK * sizeof(_Float16) + BSZ * sizeof(float);
    attn_score_kernel<<<SRC_LEN, 512, lds_bytes, stream>>>(
        src_hids, W16, decfull, v_score, scores_ws);

    softmax_kernel<<<BSZ, 256, 0, stream>>>(scores_ws, src_len, out_attn, attn_ws);

    zero_kernel<<<(BSZ * CTX_DIM) / 256, 256, 0, stream>>>(out_ctx, BSZ * CTX_DIM);
    ctx_kernel<<<dim3(BSZ, 8), 256, 0, stream>>>(src_hids, attn_ws, out_ctx);
}
